// GCN_89515708383722
// MI455X (gfx1250) — matrix operand order repaired and checked by reading, not compiled, not run on hardware
//
#include <hip/hip_runtime.h>

// ---------------------------------------------------------------------------
// GCN forward for MI455X (gfx1250, wave32, WMMA).
// GEMMs: f16 inputs, f32 accumulation via v_wmma_f32_16x16x32_f16.
// SpMM: f16 gather (keeps Y+AGG working set < 192MB L2) + f32 global atomics.
// ---------------------------------------------------------------------------

typedef __attribute__((ext_vector_type(16))) _Float16 v16h;
typedef __attribute__((ext_vector_type(8)))  float    v8f;

#define DH 256            // hidden width (fixed by problem)
#define BN_EPS 1e-3f

union Frag16 { v16h v; uint4 q[2]; };

// ------------------------------- utility -----------------------------------

__global__ void k_zero(float* __restrict__ p, long n) {
  long i = (long)blockIdx.x * blockDim.x + threadIdx.x;
  long stride = (long)gridDim.x * blockDim.x;
  for (; i < n; i += stride) p[i] = 0.0f;
}

__global__ void k_degree(const int* __restrict__ src, const int* __restrict__ dst,
                         float* __restrict__ degO, float* __restrict__ degI, int E) {
  int e = blockIdx.x * blockDim.x + threadIdx.x;
  if (e < E) {
    atomicAdd(&degO[src[e]], 1.0f);
    atomicAdd(&degI[dst[e]], 1.0f);
  }
}

__global__ void k_rsqrt_clamp(float* __restrict__ p, int n) {
  int i = blockIdx.x * blockDim.x + threadIdx.x;
  if (i < n) p[i] = rsqrtf(fmaxf(p[i], 1.0f));
}

// Xs[i][k] = f16( X[i][k] * norm_src[i] )   (row-major N x K)
__global__ void k_cvt_x(const float* __restrict__ X, const float* __restrict__ ns,
                        _Float16* __restrict__ Xs, int K, long total) {
  long idx = (long)blockIdx.x * blockDim.x + threadIdx.x;
  if (idx >= total) return;
  long i = idx / K;
  Xs[idx] = (_Float16)(X[idx] * ns[i]);
}

// Wt[n][k] = f16( W[k][n] )   (W is K x 256 row-major; Wt is 256 x K row-major)
__global__ void k_cvt_w(const float* __restrict__ W, _Float16* __restrict__ Wt, int K) {
  int idx = blockIdx.x * blockDim.x + threadIdx.x;
  if (idx >= K * DH) return;
  int k = idx / DH;
  int n = idx % DH;
  Wt[(size_t)n * K + k] = (_Float16)W[idx];
}

// ------------------------------- WMMA GEMM ---------------------------------
// Y(f16, N x 256) = Xs(f16, N x K) * Wt^T  (Wt is 256 x K row-major).
// Block = 256 threads = 8 waves. Wave w computes rows [ (bx*8+w)*16 , +16 )
// and columns [ by*64 , +64 ) as four 16x16 accumulators sharing one A frag.
// Fragment layouts follow ISA 7.12.2 (wave32):
//   A 16x32 f16 : lane L -> M = L%16 ; halves t: K = (t/8)*16 + (L/16)*8 + t%8
//   B 32x16 f16 : lane L -> N = L%16 ; halves t: K = (t/8)*16 + (L/16)*8 + t%8
//   C/D 16x16 f32: lane L -> N = L%16 ; vgpr r -> M = r + (L/16)*8
__global__ void __launch_bounds__(256)
k_gemm_wmma(const _Float16* __restrict__ Xs, const _Float16* __restrict__ Wt,
            _Float16* __restrict__ Yh, int K, int mTiles) {
  const int wave = threadIdx.x >> 5;
  const int lane = threadIdx.x & 31;
  const int mTile = blockIdx.x * 8 + wave;
  if (mTile >= mTiles) return;           // wave-uniform: EXEC stays all-ones
  const int g  = lane >> 4;              // lane group (0/1)
  const int ml = lane & 15;
  const int row0  = mTile * 16;
  const int nBase = blockIdx.y * 64;

  const _Float16* arow  = Xs + (size_t)(row0 + ml) * K;
  const _Float16* brow0 = Wt + (size_t)(nBase + ml) * K;

  v8f c[4] = {};
  for (int k0 = 0; k0 < K; k0 += 32) {
    Frag16 a;
    a.q[0] = *(const uint4*)(arow + k0 + g * 8);        // K = k0 + g*8   .. +7
    a.q[1] = *(const uint4*)(arow + k0 + 16 + g * 8);   // K = k0+16+g*8  .. +7
#pragma unroll
    for (int t = 0; t < 4; ++t) {
      Frag16 b;
      const _Float16* bp = brow0 + (size_t)(t * 16) * K + k0 + g * 8;
      b.q[0] = *(const uint4*)(bp);                     // K = k0+g*8     .. +7
      b.q[1] = *(const uint4*)(bp + 16);                // K = k0+16+g*8  .. +7
      c[t] = __builtin_amdgcn_wmma_f32_16x16x32_f16(
          /*neg_a=*/false, a.v, /*neg_b=*/false, b.v,
          /*c_mod=*/(short)0, c[t], /*reuse_a=*/false, /*reuse_b=*/false);
    }
  }

#pragma unroll
  for (int r = 0; r < 8; ++r) {
    const int row = row0 + g * 8 + r;
    _Float16* yr = Yh + (size_t)row * DH + nBase + ml;
#pragma unroll
    for (int t = 0; t < 4; ++t) yr[t * 16] = (_Float16)c[t][r];
  }
}

// ------------------------------- SpMM --------------------------------------
// One wave per edge: 32 lanes x 8 features. f16 gather (16B/lane), f32 atomics.
__global__ void k_spmm(const _Float16* __restrict__ Yh, const int* __restrict__ src,
                       const int* __restrict__ dst, float* __restrict__ agg, int E) {
  const long gid = (long)blockIdx.x * blockDim.x + threadIdx.x;
  const int e = (int)(gid >> 5);
  const int lane = (int)(gid & 31);
  if (e >= E) return;
  const int s = src[e];
  const int d = dst[e];
  union { uint4 q; _Float16 h[8]; } u;
  u.q = *(const uint4*)(Yh + (size_t)s * DH + lane * 8);
  float* ar = agg + (size_t)d * DH + lane * 8;
#pragma unroll
  for (int j = 0; j < 8; ++j) atomicAdd(ar + j, (float)u.h[j]);
}

// ----------------------- BN(eval) + ReLU + residual ------------------------
__global__ void k_finalize(const float* __restrict__ agg, const float* __restrict__ nd,
                           const float* __restrict__ bias,
                           const float* __restrict__ gamma, const float* __restrict__ beta,
                           const float* __restrict__ mean, const float* __restrict__ var,
                           const float* __restrict__ resid, float* __restrict__ Hout,
                           long total) {
  long idx = (long)blockIdx.x * blockDim.x + threadIdx.x;
  if (idx >= total) return;
  const long i = idx >> 8;       // DH == 256
  const int  j = (int)(idx & 255);
  float v = agg[idx] * nd[i] + bias[j];
  v = (v - mean[j]) * rsqrtf(var[j] + BN_EPS) * gamma[j] + beta[j];
  v = fmaxf(v, 0.0f);
  if (resid) v += resid[idx];
  Hout[idx] = v;
}

// ------------------------------- pooling + head ----------------------------
__global__ void k_pool(const float* __restrict__ H, const int* __restrict__ gids,
                       float* __restrict__ pooled, float* __restrict__ counts, int N) {
  const int i = blockIdx.x;
  if (i >= N) return;
  const int j = threadIdx.x;     // 0..255
  const int g = gids[i];
  atomicAdd(&pooled[(size_t)g * DH + j], H[(size_t)i * DH + j]);
  if (j == 0) atomicAdd(&counts[g], 1.0f);
}

__global__ void k_head(const float* __restrict__ pooled, const float* __restrict__ counts,
                       const float* __restrict__ Wp, const float* __restrict__ bp,
                       float* __restrict__ out, int G, int DOUT) {
  const int idx = blockIdx.x * blockDim.x + threadIdx.x;
  if (idx >= G * DOUT) return;
  const int g = idx / DOUT;
  const int o = idx % DOUT;
  const float inv = 1.0f / fmaxf(counts[g], 1.0f);
  float acc = 0.0f;
  for (int j = 0; j < DH; ++j) acc += pooled[(size_t)g * DH + j] * Wp[(size_t)j * DOUT + o];
  out[idx] = acc * inv + bp[o];
}

// ------------------------------- driver ------------------------------------

extern "C" void kernel_launch(void* const* d_in, const int* in_sizes, int n_in,
                              void* d_out, int out_size, void* d_ws, size_t ws_size,
                              hipStream_t stream) {
  const float* h    = (const float*)d_in[0];
  const float* W0   = (const float*)d_in[1];
  const float* b0   = (const float*)d_in[2];
  const float* W12  = (const float*)d_in[3];
  const float* b12  = (const float*)d_in[4];
  const float* bng  = (const float*)d_in[5];
  const float* bnb  = (const float*)d_in[6];
  const float* bnm  = (const float*)d_in[7];
  const float* bnv  = (const float*)d_in[8];
  const float* Wp   = (const float*)d_in[9];
  const float* bpv  = (const float*)d_in[10];
  const int*   src  = (const int*)d_in[11];
  const int*   dst  = (const int*)d_in[12];
  const int*   gids = (const int*)d_in[13];
  float* out = (float*)d_out;

  const int N    = in_sizes[13];          // 100000
  const int E    = in_sizes[11];          // 3200000
  const int DIN  = in_sizes[0] / N;       // 128
  const int DOUT = in_sizes[10];          // 64
  const int G    = out_size / DOUT;       // 128

  // ---- carve workspace (harness poisons it; everything init'd below) ----
  char* ws = (char*)d_ws;
  size_t off = 0;
  auto carve = [&](size_t bytes) -> void* {
    void* p = (void*)(ws + off);
    off += (bytes + 255) & ~(size_t)255;
    return p;
  };
  float*    normS  = (float*)carve((size_t)N * sizeof(float));
  float*    normD  = (float*)carve((size_t)N * sizeof(float));
  float*    HA     = (float*)carve((size_t)N * DH * sizeof(float));
  float*    HB     = (float*)carve((size_t)N * DH * sizeof(float));
  float*    AGG    = (float*)carve((size_t)N * DH * sizeof(float));
  _Float16* Xs     = (_Float16*)carve((size_t)N * DH * sizeof(_Float16));
  _Float16* Yh     = (_Float16*)carve((size_t)N * DH * sizeof(_Float16));
  _Float16* Wt     = (_Float16*)carve((size_t)DH * DH * sizeof(_Float16));
  float*    pooled = (float*)carve((size_t)G * DH * sizeof(float));
  float*    counts = (float*)carve((size_t)G * sizeof(float));
  (void)ws_size; (void)n_in;

  const int TB = 256;
  const long nDH = (long)N * DH;

  // ---- degree norms ----
  k_zero<<<(N + TB - 1) / TB, TB, 0, stream>>>(normS, N);
  k_zero<<<(N + TB - 1) / TB, TB, 0, stream>>>(normD, N);
  k_degree<<<(E + TB - 1) / TB, TB, 0, stream>>>(src, dst, normS, normD, E);
  k_rsqrt_clamp<<<(N + TB - 1) / TB, TB, 0, stream>>>(normS, N);
  k_rsqrt_clamp<<<(N + TB - 1) / TB, TB, 0, stream>>>(normD, N);

  // ---- three GraphConv layers ----
  struct Layer {
    const float* Xin; int K; const float* W; const float* bias;
    int bn; const float* resid; float* Hout;
  };
  const Layer L[3] = {
    { h,  DIN, W0,            b0,       0, nullptr, HA },
    { HA, DH,  W12,           b12,      1, HA,      HB },
    { HB, DH,  W12 + DH * DH, b12 + DH, 2, HB,      HA },
  };

  for (int i = 0; i < 3; ++i) {
    const int K = L[i].K;
    const long nx = (long)N * K;
    k_cvt_w<<<(K * DH + TB - 1) / TB, TB, 0, stream>>>(L[i].W, Wt, K);
    k_cvt_x<<<(int)((nx + TB - 1) / TB), TB, 0, stream>>>(L[i].Xin, normS, Xs, K, nx);
    k_zero<<<(int)((nDH + TB - 1) / TB), TB, 0, stream>>>(AGG, nDH);

    const int mTiles = N / 16;                        // N is a multiple of 16
    dim3 ggrid((mTiles + 7) / 8, DH / 64);
    k_gemm_wmma<<<ggrid, 256, 0, stream>>>(Xs, Wt, Yh, K, mTiles);

    const long spThreads = (long)E * 32;
    k_spmm<<<(int)((spThreads + TB - 1) / TB), TB, 0, stream>>>(Yh, src, dst, AGG, E);

    k_finalize<<<(int)((nDH + TB - 1) / TB), TB, 0, stream>>>(
        AGG, normD, L[i].bias,
        bng + L[i].bn * DH, bnb + L[i].bn * DH,
        bnm + L[i].bn * DH, bnv + L[i].bn * DH,
        L[i].resid, L[i].Hout, nDH);
  }

  // ---- mean pooling + prediction head ----
  k_zero<<<(G * DH + TB - 1) / TB, TB, 0, stream>>>(pooled, (long)G * DH);
  k_zero<<<1, TB, 0, stream>>>(counts, G);
  k_pool<<<N, DH, 0, stream>>>(HA, gids, pooled, counts, N);
  k_head<<<(G * DOUT + TB - 1) / TB, TB, 0, stream>>>(pooled, counts, Wp, bpv, out, G, DOUT);
}
